// RRN_89859305767511
// MI455X (gfx1250) — compile-verified, matrix operand order checked by reading, not run on hardware
//
#include <hip/hip_runtime.h>

// ---------------------------------------------------------------------------
// CDNA5 (gfx1250) wave32 WMMA bf16 implementation of the recurrent relational
// network reference. All GEMMs use v_wmma_f32_16x16x32_bf16; epilogues
// (sigmoid gates, relu, per-row L2 norm, has_class select) are f32.
// ---------------------------------------------------------------------------

typedef __bf16 bf16;
typedef __attribute__((ext_vector_type(16))) __bf16 v16bf;
typedef __attribute__((ext_vector_type(8)))  __bf16 v8bf;
typedef __attribute__((ext_vector_type(8)))  float  v8f;

#define NN   200000
#define DD   128
#define CC   64
#define RR   8
#define ERR  8192
#define QC_  8192
#define QR_  1024
#define ITERS_ 3

__device__ __forceinline__ float sigmoidf_(float x) {
    return 1.0f / (1.0f + __expf(-x));
}

// A fragment (16x32 bf16, row-major source). rowptr = start of this lane's
// matrix row (M = lane&15). Layout per ISA: elements 0..7 hold K = kbase+8*half
// .. +7, elements 8..15 hold K = kbase+16+8*half .. +7  -> two 16B loads.
__device__ __forceinline__ v16bf load_a_frag(const bf16* rowptr, int kbase, int half) {
    v8bf lo = *(const v8bf*)(rowptr + kbase + 8 * half);
    v8bf hi = *(const v8bf*)(rowptr + kbase + 16 + 8 * half);
    return __builtin_shufflevector(lo, hi, 0, 1, 2, 3, 4, 5, 6, 7,
                                   8, 9, 10, 11, 12, 13, 14, 15);
}

// B fragment from pre-swizzled weights: tile t holds 32 lanes x 16 bf16
// contiguous (512 elements), so one 32B load per lane.
__device__ __forceinline__ v16bf load_b_frag(const bf16* tiles, int tile, int lane) {
    return *(const v16bf*)(tiles + (size_t)tile * 512 + lane * 16);
}

__device__ __forceinline__ v8f wmma_bf16(v16bf a, v16bf b, v8f c) {
    return __builtin_amdgcn_wmma_f32_16x16x32_bf16(false, a, false, b,
                                                   (short)0, c, false, false);
}

// ---------------------------------------------------------------------------
// Weight pre-swizzle: f32 row-major (K x Nn) -> bf16 WMMA-B fragment order.
// grid.x = (K/32)*(Nn/16) tiles, grid.y = batch, block = 512.
// Within a tile: out[tile*512 + lane*16 + e] = W[k][n],
//   k = kt*32 + (e&7) + 8*(lane>>4) + 16*(e>>3), n = nt*16 + (lane&15).
// ---------------------------------------------------------------------------
__global__ __launch_bounds__(512) void k_swizzle_w(const float* __restrict__ W,
                                                   bf16* __restrict__ out,
                                                   int K, int Nn) {
    const int ntiles = Nn >> 4;
    const int tile = blockIdx.x;
    const int kt = tile / ntiles, nt = tile % ntiles;
    const int tid = threadIdx.x;
    const int lane = tid >> 4, e = tid & 15;
    const int k = kt * 32 + (e & 7) + 8 * (lane >> 4) + 16 * (e >> 3);
    const int n = nt * 16 + (lane & 15);
    const float* Wb = W + (size_t)blockIdx.y * K * Nn;
    bf16* ob = out + (size_t)blockIdx.y * ((size_t)(K >> 5) * ntiles * 512);
    ob[(size_t)tile * 512 + tid] = (bf16)Wb[(size_t)k * Nn + n];
}

// ---------------------------------------------------------------------------
// emb = l2norm(init_emb); write f32 + bf16 copies. 1 wave per row.
// ---------------------------------------------------------------------------
__global__ __launch_bounds__(256) void k_init_norm(const float* __restrict__ src,
                                                   float* __restrict__ cur,
                                                   bf16* __restrict__ ebf) {
    const int row = blockIdx.x * 8 + (threadIdx.x >> 5);
    const int lane = threadIdx.x & 31;
    const float4 x = *(const float4*)(src + (size_t)row * DD + lane * 4);
    float ss = x.x * x.x + x.y * x.y + x.z * x.z + x.w * x.w;
    #pragma unroll
    for (int o = 16; o >= 1; o >>= 1) ss += __shfl_xor(ss, o, 32);
    const float inv = 1.0f / fmaxf(sqrtf(ss), 1e-12f);
    float4 y;
    y.x = x.x * inv; y.y = x.y * inv; y.z = x.z * inv; y.w = x.w * inv;
    *(float4*)(cur + (size_t)row * DD + lane * 4) = y;
    bf16* ep = ebf + (size_t)row * DD + lane * 4;
    ep[0] = (bf16)y.x; ep[1] = (bf16)y.y; ep[2] = (bf16)y.z; ep[3] = (bf16)y.w;
}

// f32 -> bf16 refresh (8 elems/thread, exact multiple).
__global__ __launch_bounds__(256) void k_f32_to_bf16(const float* __restrict__ src,
                                                     bf16* __restrict__ dst) {
    const size_t base = ((size_t)blockIdx.x * blockDim.x + threadIdx.x) * 8;
    #pragma unroll
    for (int i = 0; i < 8; ++i) dst[base + i] = (bf16)src[base + i];
}

// ---------------------------------------------------------------------------
// Class update: concat = [emb | ci] (N x 192); g = sigmoid(concat@V),
// et = relu(concat@W); new = has_class ? l2norm(emb + et*g) : emb.
// One block = 16 rows; 8 waves, one 16-col n-tile each; K tiles 0..3 from
// bf16 emb, 4..5 from LDS-staged bf16 class indicators.
// ---------------------------------------------------------------------------
__global__ __launch_bounds__(256) void k_class_update(
    const float* __restrict__ cur, const bf16* __restrict__ ebf,
    const int* __restrict__ ci,
    const bf16* __restrict__ cVt, const bf16* __restrict__ cWt,
    float* __restrict__ nxt) {
    __shared__ bf16  s_ci[16][64];
    __shared__ float s_val[16][DD];
    __shared__ float s_norm[16];
    __shared__ int   s_has[16];

    const int tid = threadIdx.x;
    const int lane = tid & 31, wv = tid >> 5;
    const int half = lane >> 4;
    const int r0 = blockIdx.x * 16;

    #pragma unroll
    for (int i = 0; i < 4; ++i) {
        const int idx = tid * 4 + i;           // 0..1023 = 16 rows x 64
        const int m = idx >> 6, c = idx & 63;
        s_ci[m][c] = (bf16)(float)ci[(size_t)(r0 + m) * CC + c];
    }
    __syncthreads();

    v8f accV = {}; v8f accW = {};
    const int nt = wv;
    const int mA = lane & 15;
    #pragma unroll
    for (int kt = 0; kt < 6; ++kt) {
        v16bf a;
        if (kt < 4) a = load_a_frag(ebf + (size_t)(r0 + mA) * DD, kt * 32, half);
        else        a = load_a_frag(&s_ci[mA][0], (kt - 4) * 32, half);
        const v16bf bV = load_b_frag(cVt, kt * 8 + nt, lane);
        const v16bf bW = load_b_frag(cWt, kt * 8 + nt, lane);
        accV = wmma_bf16(a, bV, accV);
        accW = wmma_bf16(a, bW, accW);
    }

    {   // val = emb + relu(W-path) * sigmoid(V-path)
        const int n = nt * 16 + (lane & 15);
        #pragma unroll
        for (int v = 0; v < 8; ++v) {
            const int m = v + 8 * half;
            const float g  = sigmoidf_(accV[v]);
            const float et = fmaxf(accW[v], 0.0f);
            const float e  = cur[(size_t)(r0 + m) * DD + n];
            s_val[m][n] = e + et * g;
        }
    }
    __syncthreads();

    {   // per-row norm + has_class (16 threads per row)
        const int m = tid >> 4, j = tid & 15;
        float ss = 0.0f;
        #pragma unroll
        for (int c = 0; c < 8; ++c) { const float x = s_val[m][j * 8 + c]; ss += x * x; }
        int hc = 0;
        #pragma unroll
        for (int c = 0; c < 4; ++c) hc |= ((float)s_ci[m][j * 4 + c] != 0.0f) ? 1 : 0;
        #pragma unroll
        for (int o = 8; o >= 1; o >>= 1) {
            ss += __shfl_xor(ss, o, 16);
            hc |= __shfl_xor(hc, o, 16);
        }
        if (j == 0) { s_norm[m] = fmaxf(sqrtf(ss), 1e-12f); s_has[m] = hc; }
    }
    __syncthreads();

    {   // write out (has_class select)
        const int m = tid >> 4, j = tid & 15;
        const float inv = 1.0f / s_norm[m];
        const int hc = s_has[m];
        const size_t rb = (size_t)(r0 + m) * DD;
        #pragma unroll
        for (int c = 0; c < 8; ++c) {
            const int col = j * 8 + c;
            nxt[rb + col] = hc ? s_val[m][col] * inv : cur[rb + col];
        }
    }
}

// ---------------------------------------------------------------------------
// Relation update: for (r, dir) = blockIdx.y, 16 edges per block.
// es = emb[src], eo = emb[oth]; g = sigmoid(es@V1 + eo@V2);
// et = relu(es@W1 + eo@W2 + es * (eo@w)); out[src] = l2norm(es + et*g).
// subj/obj index sets are disjoint permutation slices -> conflict-free scatter.
// ---------------------------------------------------------------------------
__global__ __launch_bounds__(256) void k_rel_update(
    const float* __restrict__ cur, const bf16* __restrict__ ebf,
    const int* __restrict__ subj_idx, const int* __restrict__ obj_idx,
    const bf16* __restrict__ sV1t, const bf16* __restrict__ sV2t,
    const bf16* __restrict__ sW1t, const bf16* __restrict__ sW2t,
    const bf16* __restrict__ oV1t, const bf16* __restrict__ oV2t,
    const bf16* __restrict__ oW1t, const bf16* __restrict__ oW2t,
    const float* __restrict__ sw, const float* __restrict__ ow,
    float* __restrict__ nxt) {
    __shared__ int   s_src[16];
    __shared__ int   s_oth[16];
    __shared__ float s_edot[16];
    __shared__ float s_val[16][DD];
    __shared__ float s_norm[16];

    const int tid = threadIdx.x;
    const int lane = tid & 31, wv = tid >> 5;
    const int half = lane >> 4;
    const int combo = blockIdx.y;
    const int r = combo >> 1, dir = combo & 1;
    const size_t wOff = (size_t)r * (4 * 8 * 512);

    const bf16* V1 = (dir ? oV1t : sV1t) + wOff;
    const bf16* V2 = (dir ? oV2t : sV2t) + wOff;
    const bf16* W1 = (dir ? oW1t : sW1t) + wOff;
    const bf16* W2 = (dir ? oW2t : sW2t) + wOff;
    const float* wv_ = (dir ? ow : sw) + (size_t)r * DD;
    const int* si = (dir ? obj_idx : subj_idx) + (size_t)r * ERR;
    const int* oi = (dir ? subj_idx : obj_idx) + (size_t)r * ERR;
    const int e0 = blockIdx.x * 16;

    if (tid < 16)      s_src[tid] = si[e0 + tid];
    else if (tid < 32) s_oth[tid - 16] = oi[e0 + tid - 16];
    __syncthreads();

    {   // edot[m] = eo_row . w  (f32)
        const int m = tid >> 4, j = tid & 15;
        const size_t ob = (size_t)s_oth[m] * DD;
        float s = 0.0f;
        #pragma unroll
        for (int c = 0; c < 8; ++c) s += cur[ob + j * 8 + c] * wv_[j * 8 + c];
        #pragma unroll
        for (int o = 8; o >= 1; o >>= 1) s += __shfl_xor(s, o, 16);
        if (j == 0) s_edot[m] = s;
    }
    __syncthreads();

    v8f accV = {}; v8f accW = {};
    const int nt = wv;
    const int mA = lane & 15;
    const bf16* arow_s = ebf + (size_t)s_src[mA] * DD;
    const bf16* arow_o = ebf + (size_t)s_oth[mA] * DD;
    #pragma unroll
    for (int kt = 0; kt < 4; ++kt) {
        const v16bf a1 = load_a_frag(arow_s, kt * 32, half);
        accV = wmma_bf16(a1, load_b_frag(V1, kt * 8 + nt, lane), accV);
        accW = wmma_bf16(a1, load_b_frag(W1, kt * 8 + nt, lane), accW);
        const v16bf a2 = load_a_frag(arow_o, kt * 32, half);
        accV = wmma_bf16(a2, load_b_frag(V2, kt * 8 + nt, lane), accV);
        accW = wmma_bf16(a2, load_b_frag(W2, kt * 8 + nt, lane), accW);
    }

    {
        const int n = nt * 16 + (lane & 15);
        #pragma unroll
        for (int v = 0; v < 8; ++v) {
            const int m = v + 8 * half;
            const float esv = cur[(size_t)s_src[m] * DD + n];
            const float g   = sigmoidf_(accV[v]);
            const float et  = fmaxf(accW[v] + esv * s_edot[m], 0.0f);
            s_val[m][n] = esv + et * g;
        }
    }
    __syncthreads();

    {
        const int m = tid >> 4, j = tid & 15;
        float ss = 0.0f;
        #pragma unroll
        for (int c = 0; c < 8; ++c) { const float x = s_val[m][j * 8 + c]; ss += x * x; }
        #pragma unroll
        for (int o = 8; o >= 1; o >>= 1) ss += __shfl_xor(ss, o, 16);
        if (j == 0) s_norm[m] = fmaxf(sqrtf(ss), 1e-12f);
    }
    __syncthreads();

    {
        const int m = tid >> 4, j = tid & 15;
        const float inv = 1.0f / s_norm[m];
        const size_t rb = (size_t)s_src[m] * DD;
        #pragma unroll
        for (int c = 0; c < 8; ++c) {
            const int col = j * 8 + c;
            nxt[rb + col] = s_val[m][col] * inv;
        }
    }
}

// ---------------------------------------------------------------------------
// Class prediction head: h = relu(emb[qc]@cp_hW + hb);
// p = sigmoid(h@cp_oW + ob); out[i] = p[i, qc_cls[i]]. Two chained WMMA
// GEMMs; h staged as bf16 in LDS between them.
// ---------------------------------------------------------------------------
__global__ __launch_bounds__(256) void k_class_pred(
    const bf16* __restrict__ ebf, const int* __restrict__ qc_ind,
    const int* __restrict__ qc_cls,
    const bf16* __restrict__ hWt, const float* __restrict__ hb,
    const bf16* __restrict__ oWt, const float* __restrict__ ob,
    float* __restrict__ out) {
    __shared__ int  s_idx[16];
    __shared__ bf16 s_h[16][DD];

    const int tid = threadIdx.x;
    const int lane = tid & 31, wv = tid >> 5;
    const int half = lane >> 4;
    const int r0 = blockIdx.x * 16;

    if (tid < 16) s_idx[tid] = qc_ind[r0 + tid];
    __syncthreads();

    {   // GEMM 1: 16x128 <- gathered(16x128) @ (128x128)
        v8f acc = {};
        const int nt = wv;
        const bf16* arow = ebf + (size_t)s_idx[lane & 15] * DD;
        #pragma unroll
        for (int kt = 0; kt < 4; ++kt)
            acc = wmma_bf16(load_a_frag(arow, kt * 32, half),
                            load_b_frag(hWt, kt * 8 + nt, lane), acc);
        const int n = nt * 16 + (lane & 15);
        const float bias = hb[n];
        #pragma unroll
        for (int v = 0; v < 8; ++v)
            s_h[v + 8 * half][n] = (bf16)fmaxf(acc[v] + bias, 0.0f);
    }
    __syncthreads();

    if (wv < 4) {  // GEMM 2: 16x64 <- (16x128) @ (128x64); 4 n-tiles
        v8f acc = {};
        const int nt = wv;
        const bf16* arow = &s_h[lane & 15][0];
        #pragma unroll
        for (int kt = 0; kt < 4; ++kt)
            acc = wmma_bf16(load_a_frag(arow, kt * 32, half),
                            load_b_frag(oWt, kt * 4 + nt, lane), acc);
        const int n = nt * 16 + (lane & 15);
        const float bias = ob[n];
        #pragma unroll
        for (int v = 0; v < 8; ++v) {
            const int m = v + 8 * half;
            const float p = sigmoidf_(acc[v] + bias);
            if (qc_cls[r0 + m] == n) out[r0 + m] = p;
        }
    }
}

// ---------------------------------------------------------------------------
// Relation prediction head: cat = [emb[s] | emb[o]] (16x256);
// h = relu(cat @ rp_hW[r] + rp_hb[r]); out = sigmoid(h . rp_oW[r] + rp_ob[r]).
// ---------------------------------------------------------------------------
__global__ __launch_bounds__(256) void k_rel_pred(
    const bf16* __restrict__ ebf,
    const int* __restrict__ qr_subj, const int* __restrict__ qr_obj,
    const bf16* __restrict__ hWt, const float* __restrict__ hb,
    const float* __restrict__ oW, const float* __restrict__ ob,
    float* __restrict__ out) {
    __shared__ int   s_s[16];
    __shared__ int   s_o[16];
    __shared__ float s_h[16][DD];

    const int tid = threadIdx.x;
    const int lane = tid & 31, wv = tid >> 5;
    const int half = lane >> 4;
    const int r = blockIdx.y;
    const int e0 = blockIdx.x * 16;

    if (tid < 16)      s_s[tid] = qr_subj[(size_t)r * QR_ + e0 + tid];
    else if (tid < 32) s_o[tid - 16] = qr_obj[(size_t)r * QR_ + e0 + tid - 16];
    __syncthreads();

    {
        v8f acc = {};
        const int nt = wv;
        const bf16* as = ebf + (size_t)s_s[lane & 15] * DD;
        const bf16* ao = ebf + (size_t)s_o[lane & 15] * DD;
        const bf16* Wt = hWt + (size_t)r * (8 * 8 * 512);  // K=256 -> 8 k-tiles
        #pragma unroll
        for (int kt = 0; kt < 8; ++kt) {
            const v16bf a = (kt < 4) ? load_a_frag(as, kt * 32, half)
                                     : load_a_frag(ao, (kt - 4) * 32, half);
            acc = wmma_bf16(a, load_b_frag(Wt, kt * 8 + nt, lane), acc);
        }
        const int n = nt * 16 + (lane & 15);
        const float bias = hb[(size_t)r * DD + n];
        #pragma unroll
        for (int v = 0; v < 8; ++v)
            s_h[v + 8 * half][n] = fmaxf(acc[v] + bias, 0.0f);
    }
    __syncthreads();

    {
        const int m = tid >> 4, j = tid & 15;
        float s = 0.0f;
        #pragma unroll
        for (int c = 0; c < 8; ++c)
            s += s_h[m][j * 8 + c] * oW[(size_t)r * DD + j * 8 + c];
        #pragma unroll
        for (int o = 8; o >= 1; o >>= 1) s += __shfl_xor(s, o, 16);
        if (j == 0)
            out[QC_ + (size_t)r * QR_ + e0 + m] = sigmoidf_(s + ob[r]);
    }
}

// ---------------------------------------------------------------------------
// Host launcher
// ---------------------------------------------------------------------------
extern "C" void kernel_launch(void* const* d_in, const int* in_sizes, int n_in,
                              void* d_out, int out_size, void* d_ws, size_t ws_size,
                              hipStream_t stream) {
    (void)in_sizes; (void)n_in; (void)out_size; (void)ws_size;

    const float* init_emb = (const float*)d_in[0];
    const int*   ci       = (const int*)  d_in[1];
    const int*   subj_idx = (const int*)  d_in[2];
    const int*   obj_idx  = (const int*)  d_in[3];
    const int*   qc_ind   = (const int*)  d_in[4];
    const int*   qc_cls   = (const int*)  d_in[5];
    const int*   qr_subj  = (const int*)  d_in[6];
    const int*   qr_obj   = (const int*)  d_in[7];
    /* d_in[8] = num_iterations (device scalar); fixed to 3 per reference */
    const float* class_V  = (const float*)d_in[9];
    const float* class_W  = (const float*)d_in[10];
    const float* sV1 = (const float*)d_in[11];
    const float* sV2 = (const float*)d_in[12];
    const float* sW1 = (const float*)d_in[13];
    const float* sW2 = (const float*)d_in[14];
    const float* sw  = (const float*)d_in[15];
    const float* oV1 = (const float*)d_in[16];
    const float* oV2 = (const float*)d_in[17];
    const float* oW1 = (const float*)d_in[18];
    const float* oW2 = (const float*)d_in[19];
    const float* ow  = (const float*)d_in[20];
    const float* cp_hW = (const float*)d_in[21];
    const float* cp_hb = (const float*)d_in[22];
    const float* cp_oW = (const float*)d_in[23];
    const float* cp_ob = (const float*)d_in[24];
    const float* rp_hW = (const float*)d_in[25];
    const float* rp_hb = (const float*)d_in[26];
    const float* rp_oW = (const float*)d_in[27];
    const float* rp_ob = (const float*)d_in[28];
    float* out = (float*)d_out;

    const size_t ND = (size_t)NN * DD;   // 25.6M elements
    float* bufA = (float*)d_ws;
    float* bufB = bufA + ND;
    bf16*  ebf  = (bf16*)(bufB + ND);
    bf16*  p    = ebf + ND;
    bf16* cVt  = p; p += 6 * 8 * 512;            // class_V  192x128
    bf16* cWt  = p; p += 6 * 8 * 512;            // class_W  192x128
    const size_t rel1 = (size_t)RR * 4 * 8 * 512; // per tensor: 8 x (128x128)
    bf16* sV1t = p; p += rel1;  bf16* sV2t = p; p += rel1;
    bf16* sW1t = p; p += rel1;  bf16* sW2t = p; p += rel1;
    bf16* oV1t = p; p += rel1;  bf16* oV2t = p; p += rel1;
    bf16* oW1t = p; p += rel1;  bf16* oW2t = p; p += rel1;
    bf16* cphWt = p; p += 4 * 8 * 512;           // 128x128
    bf16* cpoWt = p; p += 4 * 4 * 512;           // 128x64
    bf16* rphWt = p; p += (size_t)RR * 8 * 8 * 512; // 8 x (256x128)

    // --- weight swizzle to WMMA fragment order (bf16) ---
    k_swizzle_w<<<dim3(48, 1), 512, 0, stream>>>(class_V, cVt, 192, 128);
    k_swizzle_w<<<dim3(48, 1), 512, 0, stream>>>(class_W, cWt, 192, 128);
    k_swizzle_w<<<dim3(32, RR), 512, 0, stream>>>(sV1, sV1t, 128, 128);
    k_swizzle_w<<<dim3(32, RR), 512, 0, stream>>>(sV2, sV2t, 128, 128);
    k_swizzle_w<<<dim3(32, RR), 512, 0, stream>>>(sW1, sW1t, 128, 128);
    k_swizzle_w<<<dim3(32, RR), 512, 0, stream>>>(sW2, sW2t, 128, 128);
    k_swizzle_w<<<dim3(32, RR), 512, 0, stream>>>(oV1, oV1t, 128, 128);
    k_swizzle_w<<<dim3(32, RR), 512, 0, stream>>>(oV2, oV2t, 128, 128);
    k_swizzle_w<<<dim3(32, RR), 512, 0, stream>>>(oW1, oW1t, 128, 128);
    k_swizzle_w<<<dim3(32, RR), 512, 0, stream>>>(oW2, oW2t, 128, 128);
    k_swizzle_w<<<dim3(32, 1), 512, 0, stream>>>(cp_hW, cphWt, 128, 128);
    k_swizzle_w<<<dim3(16, 1), 512, 0, stream>>>(cp_oW, cpoWt, 128, 64);
    k_swizzle_w<<<dim3(64, RR), 512, 0, stream>>>(rp_hW, rphWt, 256, 128);

    // --- emb = l2norm(init_emb) ---
    k_init_norm<<<NN / 8, 256, 0, stream>>>(init_emb, bufA, ebf);

    // --- message-passing iterations (ping-pong f32, refresh bf16) ---
    float* cur = bufA;
    float* nxt = bufB;
    for (int it = 0; it < ITERS_; ++it) {
        k_class_update<<<NN / 16, 256, 0, stream>>>(cur, ebf, ci, cVt, cWt, nxt);
        k_rel_update<<<dim3(ERR / 16, 2 * RR), 256, 0, stream>>>(
            cur, ebf, subj_idx, obj_idx,
            sV1t, sV2t, sW1t, sW2t, oV1t, oV2t, oW1t, oW2t, sw, ow, nxt);
        k_f32_to_bf16<<<(int)(ND / (256 * 8)), 256, 0, stream>>>(nxt, ebf);
        float* t = cur; cur = nxt; nxt = t;
    }

    // --- prediction heads ---
    k_class_pred<<<QC_ / 16, 256, 0, stream>>>(ebf, qc_ind, qc_cls,
                                               cphWt, cp_hb, cpoWt, cp_ob, out);
    k_rel_pred<<<dim3(QR_ / 16, RR), 256, 0, stream>>>(ebf, qr_subj, qr_obj,
                                                       rphWt, rp_hb, rp_oW, rp_ob, out);
}
